// TransformerBlock3D_89584427860386
// MI455X (gfx1250) — matrix operand order repaired and checked by reading, not compiled, not run on hardware
//
#include <hip/hip_runtime.h>

// ---------------------------------------------------------------------------
// TransformerBlock3D (Swin-3D style) for gfx1250 / MI455X.
// All GEMMs run on v_wmma_f32_16x16x32_f16 (f16 A/B, f32 accumulate).
// Tile indices are readfirstlane'd to SGPRs so epilogues use scalar branches.
// ---------------------------------------------------------------------------

typedef __attribute__((ext_vector_type(16))) _Float16 v16h;
typedef __attribute__((ext_vector_type(8)))  float    v8f;

#define DIMC    96
#define HEADS   3
#define HD      32
#define NWIN    64      // 4*4*4 tokens per window
#define HIDDEN  384
#define NWX     24      // 96/4
#define NWY     24      // 96/4
#define NWZ     16      // 64/4
#define N_WINDOWS (NWZ*NWY*NWX)          // 9216
#define N_TOKENS  (64ll*96*96*96)        // 56,623,104

// packed-weight offsets (in halfs) inside d_ws
#define OFF_WQ   0
#define OFF_WKV  9216
#define OFF_WP   27648
#define OFF_W1   36864
#define OFF_W2   73728

static __device__ __forceinline__ v8f zero8() {
  v8f z;
#pragma unroll
  for (int i = 0; i < 8; ++i) z[i] = 0.0f;
  return z;
}

// A-fragment (16x32 f16, M x K) from a row-major LDS buffer, row stride `ld`
// halves. Lane m=L&15; lane[4] selects the 16-wide K half;
// VGPR v holds K pair (L>>4)*16 + v*2.
static __device__ __forceinline__ v16h load_a(const _Float16* buf, int ld,
                                              int m0, int k0, int lane) {
  const int m = m0 + (lane & 15);
  const int hsel = lane >> 4;
  const _Float16* row = buf + m * ld + k0;
  v16h a;
#pragma unroll
  for (int v = 0; v < 8; ++v) {
    const int kb = (hsel << 4) + (v << 1);
    a[2 * v]     = row[kb];
    a[2 * v + 1] = row[kb + 1];
  }
  return a;
}

// B-fragment (32x16 f16, K x N) where element B[k][n] = buf[n0+n][k0+k]
// (buffer indexed [n][k], row stride `ld` halves).  Lanes 0-15: K=0..15,
// lanes 16-31: K=16..31; column n = lane&15.
static __device__ __forceinline__ v16h load_bT(const _Float16* buf, int ld,
                                               int k0, int n0, int lane) {
  const int n = n0 + (lane & 15);
  const int koff = k0 + ((lane >> 4) << 4);
  const _Float16* row = buf + n * ld + koff;
  v16h b;
#pragma unroll
  for (int v = 0; v < 8; ++v) {
    b[2 * v]     = row[2 * v];
    b[2 * v + 1] = row[2 * v + 1];
  }
  return b;
}

// B-fragment from pre-packed global weights: tile (nt*KT+kt) holds 512 halves
// in fragment order; each lane loads its 16 halves with two b128 loads.
static __device__ __forceinline__ v16h load_bp(const _Float16* pk, int kt,
                                               int nt, int KT, int lane) {
  const _Float16* p = pk + (((size_t)(nt * KT + kt)) << 9) + (lane << 4);
  union { uint4 u[2]; v16h h; } u;
  u.u[0] = *(const uint4*)p;
  u.u[1] = *(const uint4*)(p + 8);
  return u.h;
}

#define WMMA(A, B, C) \
  __builtin_amdgcn_wmma_f32_16x16x32_f16(false, (A), false, (B), (short)0, (C), false, false)

// ---------------------------------------------------------------------------
// Kernel 0: pack an f32 weight (K x N, row major) into f16 B-fragment tiles.
// grid = (K/32)*(N/16) blocks of 256 threads; 2 halves per thread.
// ---------------------------------------------------------------------------
__global__ __launch_bounds__(256) void pack_kernel(const float* __restrict__ src,
                                                   _Float16* __restrict__ dst,
                                                   int K, int N) {
  const int KT = K >> 5;
  const int tile = blockIdx.x;          // tile = nt*KT + kt
  const int nt = tile / KT;
  const int kt = tile - nt * KT;
  int e = threadIdx.x * 2;
#pragma unroll
  for (int j = 0; j < 2; ++j, ++e) {
    const int lane = (e >> 4) & 31;
    const int n = nt * 16 + (lane & 15);
    const int k = kt * 32 + (lane & 16) + (e & 15);
    dst[((size_t)tile << 9) + e] = (_Float16)src[(size_t)k * N + n];
  }
}

// ---------------------------------------------------------------------------
// Kernel 1: fused LN1 + windowed attention + out-proj + residual.
// One 256-thread (8-wave) workgroup per 4x4x4 window.  Writes y = x + attn.
// ---------------------------------------------------------------------------
__global__ __launch_bounds__(256) void attn_kernel(
    const float* __restrict__ x, const float* __restrict__ g1,
    const float* __restrict__ b1, const float* __restrict__ bq,
    const float* __restrict__ bkv, const float* __restrict__ bp,
    const _Float16* __restrict__ wpk, float* __restrict__ out) {
  __shared__ _Float16 tok[64 * 100];    // LN1(x) f16; later reused for attn O
  __shared__ _Float16 qb[64 * 100];     // Q (pre-scaled), [token][ch]
  __shared__ _Float16 kb[64 * 100];     // K, [token][ch]
  __shared__ _Float16 vt[96 * 68];      // V transposed, [ch][token]
  __shared__ float    sb[3 * 64 * 66];  // S f32 per head; also xwin staging
  __shared__ _Float16 pb[3 * 64 * 72];  // softmax P f16, [head][row][col]

  const int tid = threadIdx.x;
  const int lane = tid & 31;
  const int wave = tid >> 5;
  const int wi = blockIdx.x;
  const int dz = wi / (NWY * NWX);
  const int hy = (wi / NWX) % NWY;
  const int wx = wi % NWX;

  auto rowbase = [&](int t) -> size_t {
    const int z = dz * 4 + (t >> 4);
    const int y = hy * 4 + ((t >> 2) & 3);
    const int xx = wx * 4 + (t & 3);
    return ((size_t)(z * 96 + y) * 96 + xx) * 96;
  };

  // ---- gather window into sb (reused as xwin[64][96]) -------------------
  float* xwin = sb;
  for (int idx = tid; idx < 64 * 96; idx += 256) {
    const int t = idx / 96, c = idx - t * 96;
    xwin[t * 96 + c] = x[rowbase(t) + c];
  }
  __syncthreads();

  // ---- LayerNorm1 -> tok (f16) ------------------------------------------
  if (tid < 64) {
    const float* r = xwin + tid * 96;
    float s = 0.f, s2 = 0.f;
    for (int c = 0; c < 96; ++c) { const float v = r[c]; s += v; s2 += v * v; }
    const float mu = s * (1.0f / 96.0f);
    const float var = s2 * (1.0f / 96.0f) - mu * mu;
    const float inv = rsqrtf(var + 1e-5f);
    for (int c = 0; c < 96; ++c)
      tok[tid * 100 + c] = (_Float16)((r[c] - mu) * inv * g1[c] + b1[c]);
  }
  __syncthreads();

  // ---- QKV = h @ [wq | wkv]  (M=64, N=288, K=96) ------------------------
  const float scale = 0.17677669529663687f;  // 32^-0.5
  for (int t = wave; t < 72; t += 8) {       // 9 tiles per wave, uniform
    const int tu = __builtin_amdgcn_readfirstlane(t);   // -> SGPR
    const int nt = tu >> 2, mt = tu & 3;
    const _Float16* wsrc = (nt < 6) ? (wpk + OFF_WQ) : (wpk + OFF_WKV);
    const int ntl = (nt < 6) ? nt : nt - 6;
    // prefetch the B tiles for iteration t+8 while this tile computes
    if (tu + 8 < 72) {
      const int nn = (tu + 8) >> 2;
      const _Float16* ws2 = (nn < 6) ? (wpk + OFF_WQ) : (wpk + OFF_WKV);
      const int nl2 = (nn < 6) ? nn : nn - 6;
      __builtin_prefetch(ws2 + (((size_t)nl2 * 3) << 9) + (lane << 4), 0, 3);
    }
    v8f acc = zero8();
#pragma unroll
    for (int kt = 0; kt < 3; ++kt) {
      const v16h a = load_a(tok, 100, mt * 16, kt * 32, lane);
      const v16h b = load_bp(wsrc, kt, ntl, 3, lane);
      acc = WMMA(a, b, acc);
    }
    const int hsel = lane >> 4;
    const int nlo = lane & 15;
    if (nt < 6) {                                   // scalar branch: Q
      const int c = nt * 16 + nlo;
      const float bias = bq[c];
#pragma unroll
      for (int r = 0; r < 8; ++r)
        qb[(mt * 16 + r + 8 * hsel) * 100 + c] = (_Float16)((acc[r] + bias) * scale);
    } else if (nt < 12) {                           // scalar branch: K
      const int c = (nt - 6) * 16 + nlo;
      const float bias = bkv[c];
#pragma unroll
      for (int r = 0; r < 8; ++r)
        kb[(mt * 16 + r + 8 * hsel) * 100 + c] = (_Float16)(acc[r] + bias);
    } else {                                        // scalar branch: V (transposed)
      const int c = (nt - 12) * 16 + nlo;
      const float bias = bkv[96 + c];
#pragma unroll
      for (int r = 0; r < 8; ++r)
        vt[c * 68 + (mt * 16 + r + 8 * hsel)] = (_Float16)(acc[r] + bias);
    }
  }
  __syncthreads();

  // ---- S = Q @ K^T per head  (3 heads x 4x4 tiles, K=32) ----------------
  for (int t = wave; t < 48; t += 8) {       // 6 tiles per wave
    const int tu = __builtin_amdgcn_readfirstlane(t);
    const int head = tu >> 4, rem = tu & 15, mt = rem >> 2, nt2 = rem & 3;
    const v16h a = load_a(qb, 100, mt * 16, head * HD, lane);
    const v16h b = load_bT(kb, 100, head * HD, nt2 * 16, lane);
    v8f acc = zero8();
    acc = WMMA(a, b, acc);
    const int n = nt2 * 16 + (lane & 15);
    const int hsel = lane >> 4;
    float* sh = sb + head * 64 * 66;
#pragma unroll
    for (int r = 0; r < 8; ++r)
      sh[(mt * 16 + r + 8 * hsel) * 66 + n] = acc[r];
  }
  __syncthreads();

  // ---- softmax rows -> P (f16) ------------------------------------------
  if (tid < 192) {
    const float* row = sb + (tid / 64) * 64 * 66 + (tid % 64) * 66;
    _Float16* prow = pb + (tid / 64) * 64 * 72 + (tid % 64) * 72;
    float mx = row[0];
    for (int c = 1; c < 64; ++c) mx = fmaxf(mx, row[c]);
    float sum = 0.f;
    for (int c = 0; c < 64; ++c) sum += __expf(row[c] - mx);
    const float inv = 1.0f / sum;
    for (int c = 0; c < 64; ++c)
      prow[c] = (_Float16)(__expf(row[c] - mx) * inv);
  }
  __syncthreads();

  // ---- O = P @ V per head  (M=64, N=32, K=64) ---------------------------
  for (int t = wave; t < 24; t += 8) {       // 3 tiles per wave
    const int tu = __builtin_amdgcn_readfirstlane(t);
    const int head = tu >> 3, rem = tu & 7, mt = rem >> 1, nt2 = rem & 1;
    const _Float16* pbase = pb + head * 64 * 72;
    v8f acc = zero8();
#pragma unroll
    for (int kt = 0; kt < 2; ++kt) {
      const v16h a = load_a(pbase, 72, mt * 16, kt * 32, lane);
      const v16h b = load_bT(vt, 68, kt * 32, head * HD + nt2 * 16, lane);
      acc = WMMA(a, b, acc);
    }
    const int ncol = head * HD + nt2 * 16 + (lane & 15);
    const int hsel = lane >> 4;
#pragma unroll
    for (int r = 0; r < 8; ++r)
      tok[(mt * 16 + r + 8 * hsel) * 100 + ncol] = (_Float16)acc[r];
  }
  __syncthreads();

  // ---- out = O @ wp + bp + x  (M=64, N=96, K=96) ------------------------
  for (int t = wave; t < 24; t += 8) {       // 3 tiles per wave
    const int tu = __builtin_amdgcn_readfirstlane(t);
    const int nt = tu >> 2, mt = tu & 3;
    v8f acc = zero8();
#pragma unroll
    for (int kt = 0; kt < 3; ++kt) {
      const v16h a = load_a(tok, 100, mt * 16, kt * 32, lane);
      const v16h b = load_bp(wpk + OFF_WP, kt, nt, 3, lane);
      acc = WMMA(a, b, acc);
    }
    const int ncol = nt * 16 + (lane & 15);
    const float bias = bp[ncol];
    const int hsel = lane >> 4;
#pragma unroll
    for (int r = 0; r < 8; ++r) {
      const int m = mt * 16 + r + 8 * hsel;
      const size_t g = rowbase(m) + ncol;
      out[g] = x[g] + acc[r] + bias;
    }
  }
}

// ---------------------------------------------------------------------------
// Kernel 2: fused LN2 + MLP (96->384 GELU ->96) + residual, in place on d_out.
// One 256-thread workgroup per 64 contiguous tokens.
// ---------------------------------------------------------------------------
__global__ __launch_bounds__(256) void mlp_kernel(
    const float* __restrict__ g2, const float* __restrict__ b2,
    const _Float16* __restrict__ w1p, const float* __restrict__ bm1,
    const _Float16* __restrict__ w2p, const float* __restrict__ bm2,
    float* __restrict__ io) {
  __shared__ float    yb[64 * 100];     // y = x + attn (residual source)
  __shared__ _Float16 h2[64 * 100];     // LN2(y) f16
  __shared__ _Float16 gb[64 * 388];     // GELU(h2@w1+bm1) f16

  const int tid = threadIdx.x;
  const int lane = tid & 31;
  const int wave = tid >> 5;
  const size_t base = (size_t)blockIdx.x * (64 * 96);

  for (int idx = tid; idx < 64 * 96; idx += 256) {
    const int t = idx / 96, c = idx - t * 96;
    yb[t * 100 + c] = io[base + idx];
  }
  __syncthreads();

  if (tid < 64) {
    const float* r = yb + tid * 100;
    float s = 0.f, s2 = 0.f;
    for (int c = 0; c < 96; ++c) { const float v = r[c]; s += v; s2 += v * v; }
    const float mu = s * (1.0f / 96.0f);
    const float var = s2 * (1.0f / 96.0f) - mu * mu;
    const float inv = rsqrtf(var + 1e-5f);
    for (int c = 0; c < 96; ++c)
      h2[tid * 100 + c] = (_Float16)((r[c] - mu) * inv * g2[c] + b2[c]);
  }
  __syncthreads();

  // ---- H = GELU(h2 @ w1 + bm1)  (M=64, N=384, K=96) ---------------------
  for (int t = wave; t < 96; t += 8) {       // 12 tiles per wave
    const int tu = __builtin_amdgcn_readfirstlane(t);
    const int nt = tu >> 2, mt = tu & 3;
    if (tu + 8 < 96)  // prefetch next tile's weights
      __builtin_prefetch(w1p + (((size_t)((tu + 8) >> 2) * 3) << 9) + (lane << 4), 0, 3);
    v8f acc = zero8();
#pragma unroll
    for (int kt = 0; kt < 3; ++kt) {
      const v16h a = load_a(h2, 100, mt * 16, kt * 32, lane);
      const v16h b = load_bp(w1p, kt, nt, 3, lane);
      acc = WMMA(a, b, acc);
    }
    const int ncol = nt * 16 + (lane & 15);
    const float bias = bm1[ncol];
    const int hsel = lane >> 4;
#pragma unroll
    for (int r = 0; r < 8; ++r) {
      float v = acc[r] + bias;
      v = 0.5f * v * (1.0f + erff(v * 0.70710678118654752f));  // exact GELU
      gb[(mt * 16 + r + 8 * hsel) * 388 + ncol] = (_Float16)v;
    }
  }
  __syncthreads();

  // ---- out = y + H @ w2 + bm2  (M=64, N=96, K=384) ----------------------
  for (int t = wave; t < 24; t += 8) {       // 3 tiles per wave
    const int tu = __builtin_amdgcn_readfirstlane(t);
    const int nt = tu >> 2, mt = tu & 3;
    v8f acc = zero8();
#pragma unroll
    for (int kt = 0; kt < 12; ++kt) {
      const v16h a = load_a(gb, 388, mt * 16, kt * 32, lane);
      const v16h b = load_bp(w2p, kt, nt, 12, lane);
      acc = WMMA(a, b, acc);
    }
    const int ncol = nt * 16 + (lane & 15);
    const float bias = bm2[ncol];
    const int hsel = lane >> 4;
#pragma unroll
    for (int r = 0; r < 8; ++r) {
      const int m = mt * 16 + r + 8 * hsel;
      io[base + (size_t)m * 96 + ncol] = yb[m * 100 + ncol] + acc[r] + bias;
    }
  }
}

// ---------------------------------------------------------------------------
extern "C" void kernel_launch(void* const* d_in, const int* in_sizes, int n_in,
                              void* d_out, int out_size, void* d_ws, size_t ws_size,
                              hipStream_t stream) {
  const float* x   = (const float*)d_in[0];
  const float* g1  = (const float*)d_in[1];
  const float* b1  = (const float*)d_in[2];
  const float* wq  = (const float*)d_in[3];
  const float* bq  = (const float*)d_in[4];
  const float* wkv = (const float*)d_in[5];
  const float* bkv = (const float*)d_in[6];
  const float* wp  = (const float*)d_in[7];
  const float* bp  = (const float*)d_in[8];
  const float* g2  = (const float*)d_in[9];
  const float* b2  = (const float*)d_in[10];
  const float* w1  = (const float*)d_in[11];
  const float* bm1 = (const float*)d_in[12];
  const float* w2  = (const float*)d_in[13];
  const float* bm2 = (const float*)d_in[14];
  float* out = (float*)d_out;
  _Float16* wpk = (_Float16*)d_ws;   // 110,592 halves = 221,184 bytes

  // weight packing (rewritten every call -> deterministic across replays)
  pack_kernel<<<18, 256, 0, stream>>>(wq,  wpk + OFF_WQ,  96, 96);
  pack_kernel<<<36, 256, 0, stream>>>(wkv, wpk + OFF_WKV, 96, 192);
  pack_kernel<<<18, 256, 0, stream>>>(wp,  wpk + OFF_WP,  96, 96);
  pack_kernel<<<72, 256, 0, stream>>>(w1,  wpk + OFF_W1,  96, 384);
  pack_kernel<<<72, 256, 0, stream>>>(w2,  wpk + OFF_W2,  384, 96);

  // y = x + attn(LN1(x))      (writes d_out fully)
  attn_kernel<<<N_WINDOWS, 256, 0, stream>>>(x, g1, b1, bq, bkv, bp, wpk, out);

  // out = y + mlp(LN2(y))     (in place on d_out, block-local)
  mlp_kernel<<<(int)(N_TOKENS / 64), 256, 0, stream>>>(g2, b2, wpk + OFF_W1,
                                                       bm1, wpk + OFF_W2, bm2,
                                                       out);
}
